// SpatialTransformer_29300266893748
// MI455X (gfx1250) — matrix operand order repaired and checked by reading, not compiled, not run on hardware
//
#include <hip/hip_runtime.h>
#include <cstdint>
#include <cstddef>

// ---------- CDNA5 WMMA types ----------
typedef __attribute__((ext_vector_type(16))) _Float16 v16h;
typedef __attribute__((ext_vector_type(8)))  float    v8f;
typedef __attribute__((ext_vector_type(4)))  float    f32x4;

union AFrag {
    v16h     v;
    f32x4    f4[2];
    uint32_t u[8];
    _Float16 h[16];
};

__device__ __forceinline__ int iclamp(int v, int lo, int hi) {
    return v < lo ? lo : (v > hi ? hi : v);
}

// Padded image geometry for conv1: (64, 520, 520) f16, 2-pixel zero border.
// padded[b][y][x] = img[b][y-2][x-2] for y-2,x-2 in [0,512)
#define PIMG_W 520

// ============================================================================
// Pad + f32->f16 convert of the input image (memory-bound pre-pass).
// ============================================================================
__global__ __launch_bounds__(256) void pad_image(
    const float* __restrict__ img, _Float16* __restrict__ pimg)
{
    const int tid = blockIdx.x * 256 + threadIdx.x;
    if (tid >= PIMG_W * PIMG_W) return;
    const int y = tid / PIMG_W;
    const int x = tid % PIMG_W;
    const int b = blockIdx.y;
    float v = 0.0f;
    const int sy = y - 2, sx = x - 2;
    if ((unsigned)sy < 512u && (unsigned)sx < 512u)
        v = img[((size_t)b * 512 + sy) * 512 + sx];
    pimg[((size_t)b * PIMG_W + y) * PIMG_W + x] = (_Float16)v;
}

// ============================================================================
// Weight packing: f32 HWIO -> f16 WMMA B-fragment tiles.
// B (32x16, KxN) layout (ISA sparse-B table scaled to dense):
//   lane l: n = l%16, g = l/16 ; dword i, half h: kk = (i<4 ? 2i : 16+2(i-4)) + h + 8*g
// packed[((kt*NT + nt)*32 + lane)*8 + i]
// ============================================================================
__global__ __launch_bounds__(256) void pack_weights(
    const float* __restrict__ w, uint32_t* __restrict__ packed,
    int S, int CIN, int COUT, int NT, int K, int total)
{
    int tid = blockIdx.x * 256 + threadIdx.x;
    if (tid >= total) return;
    int i  = tid & 7;
    int l  = (tid >> 3) & 31;
    int nt = (tid >> 8) % NT;
    int kt = tid / (256 * NT);
    int n  = nt * 16 + (l & 15);
    int g  = l >> 4;
    uint32_t dw = 0;
    #pragma unroll
    for (int h = 0; h < 2; ++h) {
        int kk = ((i < 4) ? 2 * i : 16 + 2 * (i - 4)) + h + 8 * g;
        int k  = kt * 32 + kk;
        _Float16 v = (_Float16)0.0f;
        if (k < K) {
            int c = k % CIN;
            int t = k / CIN;
            int r = t / S;
            int s = t % S;
            v = (_Float16)w[((size_t)(r * S + s) * CIN + c) * COUT + n];
        }
        union { _Float16 f; unsigned short u; } cv; cv.f = v;
        dw |= (uint32_t)cv.u << (16 * h);
    }
    packed[tid] = dw;
}

// ============================================================================
// Conv1: 7x7x1->32 stride2 SAME (pad 2) from padded f16 image
//        + bias + relu + 2x2 maxpool -> act1 (64,128,128,32) f16
// Implicit GEMM: M = conv pixels (quad-ordered), K = 49 (pad 64), N = 32.
// Tap offsets are compile-time; g-half selection is a single v_cndmask, so all
// gathers are unconditional u16 loads that the compiler can clause.
// ============================================================================
__global__ __launch_bounds__(256) void conv1_wmma(
    const _Float16* __restrict__ pimg, const uint32_t* __restrict__ packedW,
    const float* __restrict__ bias, _Float16* __restrict__ actOut, int mtiles)
{
    constexpr int NT = 2, KT = 2, COUT = 32, PH = 128;
    const int lane = threadIdx.x & 31;
    const int wave = blockIdx.x * (blockDim.x >> 5) + (threadIdx.x >> 5);
    if (wave >= mtiles) return;
    const int g  = lane >> 4;
    const bool hi = (g != 0);

    // A-row pixel (quad-ordered M): m -> (pooled pixel p, quad q)
    const int mrow = (wave << 4) + (lane & 15);
    const int q  = mrow & 3;
    const int p  = mrow >> 2;
    const int px = p % PH;
    const int py = (p / PH) % PH;
    const int b  = p / (PH * PH);
    const int oy = 2 * py + (q >> 1);
    const int ox = 2 * px + (q & 1);

    // Padded-image base for this conv pixel: tap (r,s) lives at base[r*520+s]
    const _Float16* base =
        pimg + ((size_t)(b * PIMG_W) + 2 * oy) * PIMG_W + 2 * ox;

    v8f acc[NT];
    const v8f vzero = {0.f,0.f,0.f,0.f,0.f,0.f,0.f,0.f};
    #pragma unroll
    for (int nt = 0; nt < NT; ++nt) acc[nt] = vzero;

    #pragma unroll
    for (int kt = 0; kt < KT; ++kt) {
        AFrag af;
        // ISA 16-bit A layout: dword i, half h -> kk = (i<4 ? 2i : 16+2(i-4)) + 8g + h
        #pragma unroll
        for (int i = 0; i < 8; ++i) {
            #pragma unroll
            for (int h = 0; h < 2; ++h) {
                const int c  = ((i < 4) ? 2 * i : 16 + 2 * (i - 4)) + h + kt * 32;
                const int k0 = c;        // tap for lanes 0-15
                const int k1 = c + 8;    // tap for lanes 16-31
                _Float16 v;
                if (k0 < 49 && k1 < 49) {
                    const int o0 = (k0 / 7) * PIMG_W + (k0 % 7);
                    const int o1 = (k1 / 7) * PIMG_W + (k1 % 7);
                    v = base[hi ? o1 : o0];            // cndmask + clause-able load
                } else if (k0 < 49) {
                    const int o0 = (k0 / 7) * PIMG_W + (k0 % 7);
                    const _Float16 t = base[o0];       // unconditional, then select
                    v = hi ? (_Float16)0.0f : t;
                } else {
                    v = (_Float16)0.0f;                // K padding (49..63)
                }
                af.h[2 * i + h] = v;
            }
        }
        #pragma unroll
        for (int nt = 0; nt < NT; ++nt) {
            const v16h bf = *reinterpret_cast<const v16h*>(
                packedW + ((size_t)(kt * NT + nt) * 32 + lane) * 8);
            acc[nt] = __builtin_amdgcn_wmma_f32_16x16x32_f16(
                false, af.v, false, bf, (short)0, acc[nt], false, false);
        }
    }

    // Epilogue: bias + relu + 2x2 maxpool (quads live in acc vgprs 0-3 / 4-7)
    const int n0 = lane & 15;
    const int t0 = (wave << 2) + g * 2;
    #pragma unroll
    for (int nt = 0; nt < NT; ++nt) {
        const int n = nt * 16 + n0;
        const float bv = bias[n];
        float p0 = fmaxf(fmaxf(acc[nt][0], acc[nt][1]), fmaxf(acc[nt][2], acc[nt][3]));
        float p1 = fmaxf(fmaxf(acc[nt][4], acc[nt][5]), fmaxf(acc[nt][6], acc[nt][7]));
        p0 = fmaxf(p0 + bv, 0.0f);
        p1 = fmaxf(p1 + bv, 0.0f);
        actOut[(size_t)t0 * COUT + n]       = (_Float16)p0;
        actOut[(size_t)(t0 + 1) * COUT + n] = (_Float16)p1;
    }
}

// ============================================================================
// Generic conv (Cin multiple of 32, f16 NHWC input) + bias + relu + maxpool2.
// Each 32-wide k-tile = one spatial tap's (sub)channel-vector -> A fragment is
// two contiguous 16B loads per lane.
// ============================================================================
template<int R, int S, int CIN, int COUT, int IH, int PH, int PAD, int KT, int NT>
__global__ __launch_bounds__(256) void convN_wmma(
    const _Float16* __restrict__ actIn, const uint32_t* __restrict__ packedW,
    const float* __restrict__ bias, _Float16* __restrict__ actOut, int mtiles)
{
    const int lane = threadIdx.x & 31;
    const int wave = blockIdx.x * (blockDim.x >> 5) + (threadIdx.x >> 5);
    if (wave >= mtiles) return;
    const int g = lane >> 4;

    const int mrow = (wave << 4) + (lane & 15);
    const int q  = mrow & 3;
    const int p  = mrow >> 2;
    const int px = p % PH;
    const int py = (p / PH) % PH;
    const int b  = p / (PH * PH);
    const int oy = 2 * py + (q >> 1);
    const int ox = 2 * px + (q & 1);

    v8f acc[NT];
    const v8f vzero = {0.f,0.f,0.f,0.f,0.f,0.f,0.f,0.f};
    #pragma unroll
    for (int nt = 0; nt < NT; ++nt) acc[nt] = vzero;

    constexpr int CT = CIN / 32;            // k-tiles per spatial tap
    const f32x4 z4 = {0.f, 0.f, 0.f, 0.f};

    #pragma unroll
    for (int kt = 0; kt < KT; ++kt) {
        const int tap = kt / CT;
        const int cb  = (kt % CT) * 32;
        const int r   = tap / S, s = tap % S;
        const int iy  = 2 * oy - PAD + r;
        const int ix  = 2 * ox - PAD + s;

        AFrag af;
        if ((unsigned)iy < (unsigned)IH && (unsigned)ix < (unsigned)IH) {
            const _Float16* src = actIn +
                (((size_t)(b * IH + iy) * IH + ix) * CIN + cb);
            af.f4[0] = *reinterpret_cast<const f32x4*>(src + 8 * g);       // k = 8g..8g+7
            af.f4[1] = *reinterpret_cast<const f32x4*>(src + 16 + 8 * g);  // k = 16+8g..
        } else {
            af.f4[0] = z4;
            af.f4[1] = z4;
        }
        #pragma unroll
        for (int nt = 0; nt < NT; ++nt) {
            const v16h bf = *reinterpret_cast<const v16h*>(
                packedW + ((size_t)(kt * NT + nt) * 32 + lane) * 8);
            acc[nt] = __builtin_amdgcn_wmma_f32_16x16x32_f16(
                false, af.v, false, bf, (short)0, acc[nt], false, false);
        }
    }

    const int n0 = lane & 15;
    const int t0 = (wave << 2) + g * 2;
    #pragma unroll
    for (int nt = 0; nt < NT; ++nt) {
        const int n = nt * 16 + n0;
        const float bv = bias[n];
        float p0 = fmaxf(fmaxf(acc[nt][0], acc[nt][1]), fmaxf(acc[nt][2], acc[nt][3]));
        float p1 = fmaxf(fmaxf(acc[nt][4], acc[nt][5]), fmaxf(acc[nt][6], acc[nt][7]));
        p0 = fmaxf(p0 + bv, 0.0f);
        p1 = fmaxf(p1 + bv, 0.0f);
        actOut[(size_t)t0 * COUT + n]       = (_Float16)p0;
        actOut[(size_t)(t0 + 1) * COUT + n] = (_Float16)p1;
    }
}

// ============================================================================
// Global average pool over 8x8 spatial -> (64,128) f32
// ============================================================================
__global__ __launch_bounds__(128) void gap_kernel(
    const _Float16* __restrict__ act3, float* __restrict__ gap)
{
    const int b = blockIdx.x;
    const int c = threadIdx.x;  // 128 channels
    float s = 0.0f;
    #pragma unroll 4
    for (int i = 0; i < 64; ++i)
        s += (float)act3[((size_t)b * 64 + i) * 128 + c];
    gap[b * 128 + c] = s * (1.0f / 64.0f);
}

// ============================================================================
// Dense head: 128 -> relu 256 -> relu 128 -> theta 6. One block per batch row.
// ============================================================================
__global__ __launch_bounds__(256) void dense_kernel(
    const float* __restrict__ gap,
    const float* __restrict__ d1w, const float* __restrict__ d1b,
    const float* __restrict__ d2w, const float* __restrict__ d2b,
    const float* __restrict__ tw,  const float* __restrict__ tb,
    float* __restrict__ theta)
{
    __shared__ float x[128];
    __shared__ float h1[256];
    __shared__ float h2[128];
    const int b = blockIdx.x;
    const int t = threadIdx.x;

    if (t < 128) x[t] = gap[b * 128 + t];
    __syncthreads();

    {
        float s = d1b[t];
        #pragma unroll 8
        for (int k = 0; k < 128; ++k) s += x[k] * d1w[k * 256 + t];
        h1[t] = fmaxf(s, 0.0f);
    }
    __syncthreads();

    if (t < 128) {
        float s = d2b[t];
        #pragma unroll 8
        for (int k = 0; k < 256; ++k) s += h1[k] * d2w[k * 128 + t];
        h2[t] = fmaxf(s, 0.0f);
    }
    __syncthreads();

    if (t < 6) {
        float s = tb[t];
        for (int k = 0; k < 128; ++k) s += h2[k] * tw[k * 6 + t];
        theta[b * 6 + t] = s;
    }
}

// ============================================================================
// Affine grid + bilinear sampler on the original f32 image. (64,512,512,1)
// ============================================================================
__global__ __launch_bounds__(256) void sampler_kernel(
    const float* __restrict__ img, const float* __restrict__ theta,
    float* __restrict__ out)
{
    const size_t idx = (size_t)blockIdx.x * 256 + threadIdx.x; // 64*512*512
    const int x = (int)(idx & 511);
    const int y = (int)((idx >> 9) & 511);
    const int b = (int)(idx >> 18);

    const float* th = theta + b * 6;
    const float Xc = -1.0f + x * (2.0f / 511.0f);
    const float Yc = -1.0f + y * (2.0f / 511.0f);
    const float tx = th[0] * Xc + th[1] * Yc + th[2];
    const float ty = th[3] * Xc + th[4] * Yc + th[5];

    const float fx = (tx + 1.0f) * (511.0f * 0.5f);
    const float fy = (ty + 1.0f) * (511.0f * 0.5f);
    const int fx0 = (int)floorf(fx);
    const int fy0 = (int)floorf(fy);
    const int x0 = iclamp(fx0,     0, 511);
    const int x1 = iclamp(fx0 + 1, 0, 511);
    const int y0 = iclamp(fy0,     0, 511);
    const int y1 = iclamp(fy0 + 1, 0, 511);

    const float* ib = img + (size_t)b * 512 * 512;
    const float Ia  = ib[(size_t)y0 * 512 + x0];
    const float Ibv = ib[(size_t)y1 * 512 + x0];
    const float Ic  = ib[(size_t)y0 * 512 + x1];
    const float Id  = ib[(size_t)y1 * 512 + x1];

    const float x0f = (float)x0, x1f = (float)x1;
    const float y0f = (float)y0, y1f = (float)y1;
    const float wa = (x1f - fx) * (y1f - fy);
    const float wb = (x1f - fx) * (fy - y0f);
    const float wc = (fx - x0f) * (y1f - fy);
    const float wd = (fx - x0f) * (fy - y0f);

    out[idx] = wa * Ia + wb * Ibv + wc * Ic + wd * Id;
}

// ============================================================================
// Host launcher
// ============================================================================
extern "C" void kernel_launch(void* const* d_in, const int* in_sizes, int n_in,
                              void* d_out, int out_size, void* d_ws, size_t ws_size,
                              hipStream_t stream)
{
    const float* image = (const float*)d_in[0];
    const float* w1  = (const float*)d_in[1];
    const float* b1  = (const float*)d_in[2];
    const float* w2  = (const float*)d_in[3];
    const float* b2  = (const float*)d_in[4];
    const float* w3  = (const float*)d_in[5];
    const float* b3  = (const float*)d_in[6];
    const float* d1w = (const float*)d_in[7];
    const float* d1b = (const float*)d_in[8];
    const float* d2w = (const float*)d_in[9];
    const float* d2b = (const float*)d_in[10];
    const float* tw  = (const float*)d_in[11];
    const float* tb  = (const float*)d_in[12];
    float* out = (float*)d_out;

    char* ws = (char*)d_ws;
    size_t o = 0;
    auto alloc = [&](size_t bytes) {
        size_t r = o;
        o += (bytes + 255) & ~(size_t)255;
        return r;
    };

    uint32_t* pw1 = (uint32_t*)(ws + alloc((size_t)2  * 2 * 1024));
    uint32_t* pw2 = (uint32_t*)(ws + alloc((size_t)25 * 4 * 1024));
    uint32_t* pw3 = (uint32_t*)(ws + alloc((size_t)18 * 8 * 1024));
    _Float16* pimg = (_Float16*)(ws + alloc((size_t)64 * PIMG_W * PIMG_W * 2));
    _Float16* act1 = (_Float16*)(ws + alloc((size_t)64 * 128 * 128 * 32 * 2));
    _Float16* act2 = (_Float16*)(ws + alloc((size_t)64 * 32 * 32 * 64 * 2));
    _Float16* act3 = (_Float16*)(ws + alloc((size_t)64 * 8 * 8 * 128 * 2));
    float* gap   = (float*)(ws + alloc((size_t)64 * 128 * 4));
    float* theta = (float*)(ws + alloc((size_t)64 * 6 * 4));
    (void)o; (void)ws_size; (void)in_sizes; (void)n_in; (void)out_size;

    // Pack weights into WMMA B-fragment tiles (f16)
    {
        int tot1 = 2 * 2 * 32 * 8;
        pack_weights<<<(tot1 + 255) / 256, 256, 0, stream>>>(w1, pw1, 7, 1, 32, 2, 49, tot1);
        int tot2 = 25 * 4 * 32 * 8;
        pack_weights<<<(tot2 + 255) / 256, 256, 0, stream>>>(w2, pw2, 5, 32, 64, 4, 800, tot2);
        int tot3 = 18 * 8 * 32 * 8;
        pack_weights<<<(tot3 + 255) / 256, 256, 0, stream>>>(w3, pw3, 3, 64, 128, 8, 576, tot3);
    }

    // Pad + convert image for conv1's gather
    {
        dim3 grid((PIMG_W * PIMG_W + 255) / 256, 64);
        pad_image<<<grid, 256, 0, stream>>>(image, pimg);
    }

    // Conv1: mtiles = 64*128*128/4 = 262144 (8 waves / block)
    conv1_wmma<<<262144 / 8, 256, 0, stream>>>(pimg, pw1, b1, act1, 262144);

    // Conv2: (64,128,128,32) -> (64,32,32,64); mtiles = 16384
    convN_wmma<5, 5, 32, 64, 128, 32, 1, 25, 4>
        <<<16384 / 8, 256, 0, stream>>>(act1, pw2, b2, act2, 16384);

    // Conv3: (64,32,32,64) -> (64,8,8,128); mtiles = 1024
    convN_wmma<3, 3, 64, 128, 32, 8, 0, 18, 8>
        <<<1024 / 8, 256, 0, stream>>>(act2, pw3, b3, act3, 1024);

    // GAP -> dense head -> theta
    gap_kernel<<<64, 128, 0, stream>>>(act3, gap);
    dense_kernel<<<64, 256, 0, stream>>>(gap, d1w, d1b, d2w, d2b, tw, tb, theta);

    // Affine grid + bilinear sample (64*512*512 threads)
    sampler_kernel<<<(64 * 512 * 512) / 256, 256, 0, stream>>>(image, theta, out);
}
